// SelfAttention_25366076850538
// MI455X (gfx1250) — compile-verified, matrix operand order checked by reading, not run on hardware
//
#include <hip/hip_runtime.h>

typedef unsigned short u16;
typedef __attribute__((ext_vector_type(16))) __bf16 v16bf;
typedef __attribute__((ext_vector_type(8)))  float  v8f;

constexpr int BATCH = 8;
constexpr int SEQ   = 2048;
constexpr int DIM   = 1024;   // D_IN == D_OUT

__device__ __forceinline__ u16 f32_to_bf16(float f) {
  union { float f; unsigned u; } v; v.f = f;
  unsigned r = v.u + 0x7FFFu + ((v.u >> 16) & 1u);   // round-to-nearest-even
  return (u16)(r >> 16);
}

union FragB16 { unsigned u32[8]; uint4 q[2]; v16bf v; };

// CDNA5 async copy: global -> LDS, 16B per lane, tracked by ASYNCcnt.
#define ASYNC_LD_B128(ldsoff, gptr)                                           \
  asm volatile("global_load_async_to_lds_b128 %0, %1, off"                    \
               :: "v"(ldsoff), "v"(gptr) : "memory")

// C[M x N] = A[M x K] * Bsrc[N x K]^T   (both operands bf16 bits, row-major, ld == K)
// mode 0: Cf[m*ldOut+n] = v (f32)
// mode 1: Cb[m*ldOut+n] = bf16(v)
// mode 2: Cb[batch(m)][n][s(m)] = bf16(v)   (per-batch transposed store for V^T)
__device__ __forceinline__ void gemm_body(
    const u16* __restrict__ A, const u16* __restrict__ Bsrc,
    float* __restrict__ Cf, u16* __restrict__ Cb,
    const float* __restrict__ bias, float scale,
    int K, int ldOut, int bm, int bn, int mode)
{
  // double-buffered 128x32 bf16 tiles, rows padded to 40 halves (80B)
  __shared__ __align__(16) u16 As[2][128 * 40];
  __shared__ __align__(16) u16 Bs[2][128 * 40];
  const int tid  = threadIdx.x;
  const int lane = tid & 31;
  const int wave = tid >> 5;
  const int half = lane >> 4;       // 0: lanes 0-15, 1: lanes 16-31
  const int r16  = lane & 15;
  const int wm = (wave & 3) * 32;   // wave tile origin in block (M)
  const int wn = (wave >> 2) * 64;  // wave tile origin in block (N)

  // staging: 512 16B chunks per 8KB tile; this thread owns chunks tid, tid+256
  const int r0 = tid >> 2,          c0 = (tid & 3) * 8;
  const int r1 = (tid + 256) >> 2;  // column is identical for chunk tid+256
  const u16* gA0 = A    + (size_t)(bm + r0) * K + c0;
  const u16* gA1 = A    + (size_t)(bm + r1) * K + c0;
  const u16* gB0 = Bsrc + (size_t)(bn + r0) * K + c0;
  const u16* gB1 = Bsrc + (size_t)(bn + r1) * K + c0;
  unsigned lA0[2], lA1[2], lB0[2], lB1[2];
  for (int p = 0; p < 2; ++p) {
    // low 32 bits of the generic pointer == LDS byte offset (AS3 aperture layout)
    lA0[p] = (unsigned)(unsigned long long)(void*)&As[p][r0 * 40 + c0];
    lA1[p] = (unsigned)(unsigned long long)(void*)&As[p][r1 * 40 + c0];
    lB0[p] = (unsigned)(unsigned long long)(void*)&Bs[p][r0 * 40 + c0];
    lB1[p] = (unsigned)(unsigned long long)(void*)&Bs[p][r1 * 40 + c0];
  }
  auto issue = [&](int p, int k0) {
    ASYNC_LD_B128(lA0[p], gA0 + k0);
    ASYNC_LD_B128(lA1[p], gA1 + k0);
    ASYNC_LD_B128(lB0[p], gB0 + k0);
    ASYNC_LD_B128(lB1[p], gB1 + k0);
  };

  v8f acc[2][4];
  for (int t = 0; t < 2; ++t)
    for (int u = 0; u < 4; ++u)
      for (int e = 0; e < 8; ++e) acc[t][u][e] = 0.0f;

  issue(0, 0);
  int par = 0;
  for (int k0 = 0; k0 < K; k0 += 32) {
    const bool more = (k0 + 32) < K;
    if (more) {
      issue(par ^ 1, k0 + 32);                       // prefetch next tile
      asm volatile("s_wait_asynccnt 0x4" ::: "memory");  // retire current tile's 4
    } else {
      asm volatile("s_wait_asynccnt 0x0" ::: "memory");
    }
    __syncthreads();                                 // all waves' tiles landed

    // 16-bit A/B fragment layout: lane holds row (A) / col (B) = (lane&15);
    // k = {8*half .. 8*half+7} then {16+8*half .. 16+8*half+7}
    FragB16 fa[2], fb[4];
    for (int t = 0; t < 2; ++t) {
      const u16* rp = &As[par][(wm + 16 * t + r16) * 40];
      fa[t].q[0] = *(const uint4*)(rp + 8 * half);
      fa[t].q[1] = *(const uint4*)(rp + 16 + 8 * half);
    }
    for (int u = 0; u < 4; ++u) {
      const u16* rp = &Bs[par][(wn + 16 * u + r16) * 40];
      fb[u].q[0] = *(const uint4*)(rp + 8 * half);
      fb[u].q[1] = *(const uint4*)(rp + 16 + 8 * half);
    }
    for (int t = 0; t < 2; ++t)
      for (int u = 0; u < 4; ++u)
        acc[t][u] = __builtin_amdgcn_wmma_f32_16x16x32_bf16(
            false, fa[t].v, false, fb[u].v, (short)0, acc[t][u], false, false);

    __syncthreads();                                 // reads done; buffer reusable
    par ^= 1;
  }

  // epilogue: C element (VGPR e, lane) -> row = e + 8*half, col = lane&15
  for (int t = 0; t < 2; ++t) {
    for (int u = 0; u < 4; ++u) {
      const int gn = bn + wn + 16 * u + r16;
      const float bval = bias ? bias[gn] : 0.0f;
      for (int e = 0; e < 8; ++e) {
        const int gm = bm + wm + 16 * t + 8 * half + e;
        const float v = (acc[t][u][e] + bval) * scale;
        if (mode == 0) {
          Cf[(size_t)gm * ldOut + gn] = v;
        } else if (mode == 1) {
          Cb[(size_t)gm * ldOut + gn] = f32_to_bf16(v);
        } else {  // per-batch transposed store: Vt[b][n][s]
          const int bb = gm >> 11, s = gm & (SEQ - 1);
          Cb[(size_t)bb * DIM * SEQ + (size_t)gn * SEQ + s] = f32_to_bf16(v);
        }
      }
    }
  }
}

// ---- kernels -------------------------------------------------------------

__global__ __launch_bounds__(256) void k_cvt(const float* __restrict__ in,
                                             u16* __restrict__ out, int n4) {
  const int i = blockIdx.x * 256 + threadIdx.x;
  if (i >= n4) return;
  const float4 f = ((const float4*)in)[i];
  const unsigned lo = (unsigned)f32_to_bf16(f.x) | ((unsigned)f32_to_bf16(f.y) << 16);
  const unsigned hi = (unsigned)f32_to_bf16(f.z) | ((unsigned)f32_to_bf16(f.w) << 16);
  ((uint2*)out)[i] = make_uint2(lo, hi);
}

__global__ __launch_bounds__(256) void k_proj(
    const u16* __restrict__ xh, const u16* __restrict__ Wh,
    const float* __restrict__ bk, const float* __restrict__ bq,
    const float* __restrict__ bvv,
    u16* __restrict__ Kb, u16* __restrict__ Qb, u16* __restrict__ Vt) {
  const int z = blockIdx.z;
  const u16* W = Wh + (size_t)z * DIM * DIM;
  const float* bias = (z == 0) ? bk : ((z == 1) ? bq : bvv);
  u16* outb = (z == 0) ? Kb : ((z == 1) ? Qb : Vt);
  const float scale = (z == 0) ? 0.03125f : 1.0f;   // fold 1/sqrt(1024) into keys
  const int mode = (z == 2) ? 2 : 1;
  gemm_body(xh, W, nullptr, outb, bias, scale, DIM, DIM,
            blockIdx.y * 128, blockIdx.x * 128, mode);
}

__global__ __launch_bounds__(256) void k_scores(
    const u16* __restrict__ Kb, const u16* __restrict__ Qb,
    float* __restrict__ raw) {
  const int b = blockIdx.z;
  gemm_body(Kb + (size_t)b * SEQ * DIM, Qb + (size_t)b * SEQ * DIM,
            raw + (size_t)b * SEQ * SEQ, nullptr, nullptr, 1.0f,
            DIM, SEQ, blockIdx.y * 128, blockIdx.x * 128, 0);
}

__global__ __launch_bounds__(256) void k_softmax(
    const float* __restrict__ raw, u16* __restrict__ P) {
  __shared__ float red[256];
  const int row = blockIdx.x;           // b*SEQ + j
  const int j = row & (SEQ - 1);
  const int tid = threadIdx.x;
  const float* rp = raw + (size_t)row * SEQ;
  float r[8];
  float mx = -3.4e38f;
  for (int q = 0; q < 8; ++q) { r[q] = rp[tid + 256 * q]; mx = fmaxf(mx, r[q]); }
  red[tid] = mx; __syncthreads();
  for (int s = 128; s > 0; s >>= 1) {
    if (tid < s) red[tid] = fmaxf(red[tid], red[tid + s]);
    __syncthreads();
  }
  mx = red[0]; __syncthreads();
  float e[8], sum = 0.0f;
  for (int q = 0; q < 8; ++q) { e[q] = __expf(r[q] - mx); sum += e[q]; }
  red[tid] = sum; __syncthreads();
  for (int s = 128; s > 0; s >>= 1) {
    if (tid < s) red[tid] += red[tid + s];
    __syncthreads();
  }
  const float inv = 1.0f / red[0];
  u16* pp = P + (size_t)row * SEQ;
  for (int q = 0; q < 8; ++q) {
    const int l = tid + 256 * q;
    float v = e[q] * inv;
    if (l < j) v = -v;                  // post-softmax sign mask
    pp[l] = f32_to_bf16(v);
  }
}

__global__ __launch_bounds__(256) void k_out(
    const u16* __restrict__ P, const u16* __restrict__ Vt,
    float* __restrict__ out) {
  const int b = blockIdx.z;
  gemm_body(P + (size_t)b * SEQ * SEQ, Vt + (size_t)b * DIM * SEQ,
            out + (size_t)b * SEQ * DIM, nullptr, nullptr, 1.0f,
            SEQ, DIM, blockIdx.y * 128, blockIdx.x * 128, 0);
}

// ---- host launch ---------------------------------------------------------

extern "C" void kernel_launch(void* const* d_in, const int* in_sizes, int n_in,
                              void* d_out, int out_size, void* d_ws, size_t ws_size,
                              hipStream_t stream) {
  const float* x  = (const float*)d_in[0];
  const float* Wk = (const float*)d_in[1];
  const float* bk = (const float*)d_in[2];
  const float* Wq = (const float*)d_in[3];
  const float* bq = (const float*)d_in[4];
  const float* Wv = (const float*)d_in[5];
  const float* bv = (const float*)d_in[6];
  float* out = (float*)d_out;

  char* ws = (char*)d_ws;
  size_t off = 0;
  auto carve = [&](size_t bytes) { size_t o = off; off += (bytes + 255) & ~(size_t)255; return o; };
  const size_t MTOT = (size_t)BATCH * SEQ;                      // 16384 rows
  u16*   xh  = (u16*)(ws + carve(MTOT * DIM * 2));              // 32 MB
  u16*   Wh  = (u16*)(ws + carve((size_t)3 * DIM * DIM * 2));   // 6 MB
  u16*   Kb  = (u16*)(ws + carve(MTOT * DIM * 2));              // 32 MB
  u16*   Qb  = (u16*)(ws + carve(MTOT * DIM * 2));              // 32 MB
  u16*   Vt  = (u16*)(ws + carve(MTOT * DIM * 2));              // 32 MB [B][D][S]
  float* raw = (float*)(ws + carve((size_t)BATCH * SEQ * SEQ * 4)); // 128 MB
  u16*   P   = (u16*)(ws + carve((size_t)BATCH * SEQ * SEQ * 2));   // 64 MB

  // 1) f32 -> bf16 conversions
  {
    const int n4x = (int)(MTOT * DIM / 4);
    k_cvt<<<dim3((n4x + 255) / 256), dim3(256), 0, stream>>>(x, xh, n4x);
    const int n4w = DIM * DIM / 4;
    k_cvt<<<dim3((n4w + 255) / 256), dim3(256), 0, stream>>>(Wk, Wh, n4w);
    k_cvt<<<dim3((n4w + 255) / 256), dim3(256), 0, stream>>>(Wq, Wh + (size_t)DIM * DIM, n4w);
    k_cvt<<<dim3((n4w + 255) / 256), dim3(256), 0, stream>>>(Wv, Wh + (size_t)2 * DIM * DIM, n4w);
  }
  // 2) projections: Kb (scaled), Qb, Vt
  k_proj<<<dim3(DIM / 128, MTOT / 128, 3), dim3(256), 0, stream>>>(
      xh, Wh, bk, bq, bv, Kb, Qb, Vt);
  // 3) raw scores = Kb . Qb^T  (scale already folded into Kb)
  k_scores<<<dim3(SEQ / 128, SEQ / 128, BATCH), dim3(256), 0, stream>>>(Kb, Qb, raw);
  // 4) softmax over l + sign mask -> P (bf16)
  k_softmax<<<dim3(BATCH * SEQ), dim3(256), 0, stream>>>(raw, P);
  // 5) out = P . V   (via Vt)
  k_out<<<dim3(DIM / 128, SEQ / 128, BATCH), dim3(256), 0, stream>>>(P, Vt, out);
}
